// RelativeMultiHeadAttention_53558242181862
// MI455X (gfx1250) — matrix operand order repaired and checked by reading, not compiled, not run on hardware
//
#include <hip/hip_runtime.h>

#define DEVINL __device__ __forceinline__

typedef unsigned short u16;
typedef __attribute__((ext_vector_type(16))) __bf16 v16bf;
typedef __attribute__((ext_vector_type(8)))  __bf16 v8bf;
typedef __attribute__((ext_vector_type(8)))  float  v8f;
typedef __attribute__((ext_vector_type(4)))  unsigned int u32x4;
typedef __attribute__((ext_vector_type(8)))  int i32x8;
typedef __attribute__((ext_vector_type(4)))  int i32x4;

constexpr int Bb = 4, Ls = 1024, Dd = 1024, Hh = 16, HD = 64;

DEVINL u16 f2bf(float f) {              // round-to-nearest-even f32 -> bf16
  unsigned u = __float_as_uint(f);
  u += 0x7FFFu + ((u >> 16) & 1u);
  return (u16)(u >> 16);
}

DEVINL v8f vzero8() {
  v8f v;
#pragma unroll
  for (int i = 0; i < 8; ++i) v[i] = 0.0f;
  return v;
}

DEVINL v8bf ld8(const u16* p) { return *reinterpret_cast<const v8bf*>(p); }

DEVINL v16bf ldpair(const u16* p0, const u16* p1) {
  v8bf a = ld8(p0), b = ld8(p1);
  v16bf r;
#pragma unroll
  for (int i = 0; i < 8; ++i) { r[i] = a[i]; r[i + 8] = b[i]; }
  return r;
}

DEVINL v8f wmma_bf16(v16bf a, v16bf b, v8f c) {
  return __builtin_amdgcn_wmma_f32_16x16x32_bf16(false, a, false, b, (short)0, c,
                                                 false, false);
}

// ---- Tensor Data Mover: 2D tile (tile_x x tile_y, 2-byte elems) global -> LDS,
// ---- with LDS padding of 4 DWORDs every 16 DWORDs (64B row -> 80B LDS stride).
DEVINL void tdm_load_2d(unsigned lds_off, const u16* gptr,
                        unsigned tile_x, unsigned tile_y, unsigned row_stride) {
  unsigned long long ga = (unsigned long long)(__SIZE_TYPE__)gptr;
  u32x4 g0;
  g0[0] = 1u;                                              // count = 1 valid descriptor
  g0[1] = lds_off;                                         // lds_addr (bytes)
  g0[2] = (unsigned)(ga & 0xFFFFFFFFu);                    // global_addr[31:0]
  g0[3] = (unsigned)((ga >> 32) & 0x01FFFFFFu) | (2u << 30); // addr[56:32] | type=2
  i32x8 g1;
  g1[0] = (int)((1u << 16) |                               // data_size = 2 bytes
                (1u << 20) |                               // pad_enable
                (3u << 22) |                               // pad_interval: 16 DWORDs
                (3u << 25));                               // pad_amount: 4 DWORDs
  g1[1] = (int)((row_stride & 0xFFFFu) << 16);             // tensor_dim0[15:0]
  g1[2] = (int)(((row_stride >> 16) & 0xFFFFu) |           // tensor_dim0[31:16]
                (4096u << 16));                            // tensor_dim1[15:0]
  g1[3] = (int)((tile_x & 0xFFFFu) << 16);                 // tile_dim0
  g1[4] = (int)(tile_y & 0xFFFFu);                         // tile_dim1 (tile_dim2 = 0)
  g1[5] = (int)row_stride;                                 // tensor_dim0_stride[31:0]
  g1[6] = 0;
  g1[7] = 0;
  i32x4 z4 = {0, 0, 0, 0};
#if __clang_major__ >= 23
  i32x8 z8 = {0, 0, 0, 0, 0, 0, 0, 0};
  __builtin_amdgcn_tensor_load_to_lds(g0, g1, z4, z4, z8, 0);
#else
  __builtin_amdgcn_tensor_load_to_lds(g0, g1, z4, z4, 0);
#endif
}

// ---------------- f32 -> bf16 conversion / repack kernels ----------------

__global__ void cvt_copy_kernel(const float* __restrict__ s, u16* __restrict__ d, int n) {
  int i = blockIdx.x * blockDim.x + threadIdx.x;
  if (i < n) d[i] = f2bf(s[i]);
}

// dst[cols, rows] = transpose of src[rows, cols]
__global__ void cvt_transpose_kernel(const float* __restrict__ s, u16* __restrict__ d,
                                     int rows, int cols) {
  int i = blockIdx.x * blockDim.x + threadIdx.x;
  if (i < rows * cols) {
    int n = i / rows, k = i % rows;
    d[i] = f2bf(s[(size_t)k * cols + n]);
  }
}

__global__ void cvt_rel_kernel(const float* __restrict__ s, u16* __restrict__ d) {
  int i = blockIdx.x * blockDim.x + threadIdx.x;   // exactly 2048*64 threads
  int row = i >> 6; if (row > 2046) row = 2046;    // pad row 2047 (never used)
  d[i] = f2bf(s[(size_t)row * 64 + (i & 63)]);
}

// ---------------- QKV projection GEMM: [4096,1024] x [1024,3072] ----------------
// Block of 8 waves computes 128x64; shared B panel TDM-staged into LDS,
// double-buffered with s_wait_tensorcnt + barrier. A register double-buffered.
// B tiles batched into registers before the WMMA burst (progressive ds waits).

__global__ __launch_bounds__(256) void gemm_qkv_kernel(
    const u16* __restrict__ xb, const u16* __restrict__ wT,
    u16* __restrict__ qb, u16* __restrict__ kb, u16* __restrict__ vtb) {
  alignas(16) __shared__ u16 bpanel[2][64 * 40];   // 80B/row (TDM-padded), 2 buffers
  const int tid = threadIdx.x;
  const int wv = tid >> 5, lane = tid & 31, half = lane >> 4, l15 = lane & 15;
  const int m0 = (blockIdx.x / 48) * 128 + wv * 16;
  const int n0 = (blockIdx.x % 48) * 64;

  if (tid == 0)   // TDM issue is wave-level (EXEC-independent): exactly one issue
    tdm_load_2d((unsigned)(__SIZE_TYPE__)&bpanel[0][0], wT + (size_t)n0 * Dd, 32, 64, Dd);

  v8f c[4];
#pragma unroll
  for (int i = 0; i < 4; ++i) c[i] = vzero8();

  const u16* arow = xb + (size_t)(m0 + l15) * Dd;
  const u16* ap0 = arow + 16 * half;
  v16bf a = ldpair(ap0, ap0 + 8);

  for (int k = 0, p = 0; k < Dd; k += 32, p ^= 1) {
    if (tid == 0) __builtin_amdgcn_s_wait_tensorcnt(0);
    __syncthreads();                         // buf[p] ready; buf[p^1] fully consumed
    if (tid == 0 && k + 32 < Dd)
      tdm_load_2d((unsigned)(__SIZE_TYPE__)&bpanel[p ^ 1][0],
                  wT + (size_t)n0 * Dd + (k + 32), 32, 64, Dd);
    v16bf an = a;
    if (k + 32 < Dd) {                       // prefetch next A into registers
      const u16* ap = arow + (k + 32) + 16 * half;
      __builtin_prefetch(ap + 32, 0, 0);
      an = ldpair(ap, ap + 8);
    }
    const u16* bb = &bpanel[p][0];
    v16bf bt[4];
#pragma unroll
    for (int nt = 0; nt < 4; ++nt) {         // batch all B tiles first
      const u16* bp = bb + (nt * 16 + l15) * 40 + 16 * half;
      bt[nt] = ldpair(bp, bp + 8);
    }
#pragma unroll
    for (int nt = 0; nt < 4; ++nt)           // then back-to-back WMMAs
      c[nt] = wmma_bf16(a, bt[nt], c[nt]);
    a = an;
  }
#pragma unroll
  for (int nt = 0; nt < 4; ++nt) {
#pragma unroll
    for (int r = 0; r < 8; ++r) {
      const float val = c[nt][r];
      const int row = m0 + r + 8 * half;
      const int col = n0 + nt * 16 + l15;
      const int bb2 = row >> 10, l = row & 1023;
      const int sel = col >> 10, rem = col & 1023;
      const int hh = rem >> 6, d = rem & 63;
      const size_t bh = (size_t)bb2 * Hh + hh;
      if (sel == 0)      qb[(bh * Ls + l) * HD + d] = f2bf(val * 0.125f);  // 1/sqrt(64)
      else if (sel == 1) kb[(bh * Ls + l) * HD + d] = f2bf(val);
      else               vtb[(bh * HD + d) * Ls + l] = f2bf(val);          // V^T
    }
  }
}

// ---------------- fused flash attention with relative-position skew ----------------
// One wave per 16-row query tile, 32 keys per iteration.

__global__ __launch_bounds__(128) void attn_kernel(
    const u16* __restrict__ qb, const u16* __restrict__ kb,
    const u16* __restrict__ vtb, const u16* __restrict__ relb,
    u16* __restrict__ attnb) {
  alignas(16) __shared__ u16 pbuf[4][16 * 40];   // P tile staging (stride 40, 16B aligned)
  const int tid = threadIdx.x;
  const int wv = tid >> 5, lane = tid & 31, half = lane >> 4, l15 = lane & 15;

  const int bh = blockIdx.x >> 4;                        // b*16 + h
  const int l0 = ((blockIdx.x & 15) << 6) + (wv << 4);   // query tile start
  const int b = bh >> 4, h = bh & 15;

  // Q tile as WMMA A-operand, both K-steps (HD = 64)
  const u16* qrow = qb + ((size_t)bh * Ls + l0 + l15) * HD;
  v16bf aq[2];
#pragma unroll
  for (int ds = 0; ds < 2; ++ds) {
    const u16* p = qrow + ds * 32 + 16 * half;
    aq[ds] = ldpair(p, p + 8);
  }

  v8f o[4];
#pragma unroll
  for (int dt = 0; dt < 4; ++dt) o[dt] = vzero8();
  float m[8], lsum[8];
#pragma unroll
  for (int r = 0; r < 8; ++r) { m[r] = -1e30f; lsum[r] = 0.0f; }

  const u16* kbase = kb + (size_t)bh * Ls * HD;
  const u16* vbase = vtb + (size_t)bh * HD * Ls;
  u16* pb = &pbuf[wv][0];

  for (int k0 = 0; k0 < Ls; k0 += 32) {
    // ---- content scores S = Q K^T: batch 4 B tiles, then 4 WMMAs ----
    v16bf kB[4];
#pragma unroll
    for (int ds = 0; ds < 2; ++ds) {
      const u16* kr0 = kbase + (size_t)(k0 + l15) * HD + ds * 32 + 16 * half;
      kB[ds] = ldpair(kr0, kr0 + 8);
      const u16* kr1 = kbase + (size_t)(k0 + 16 + l15) * HD + ds * 32 + 16 * half;
      kB[2 + ds] = ldpair(kr1, kr1 + 8);
    }
    v8f s0 = vzero8(), s1 = vzero8();
    s0 = wmma_bf16(aq[0], kB[0], s0);
    s0 = wmma_bf16(aq[1], kB[1], s0);
    s1 = wmma_bf16(aq[0], kB[2], s1);
    s1 = wmma_bf16(aq[1], kB[3], s1);

    // ---- rel-pos window E[i,jp] = q_i . R[pbase+jp], jp in [0,48) ----
    const int pbase = k0 - l0 + 1008;    // (k0 - l0) + 1023 - 15
    v16bf rB[6];
#pragma unroll
    for (int et = 0; et < 3; ++et) {
      int prow = pbase + et * 16 + l15;
      prow = prow < 0 ? 0 : (prow > 2046 ? 2046 : prow);   // clamp unused edges
#pragma unroll
      for (int ds = 0; ds < 2; ++ds) {
        const u16* rr = relb + (size_t)prow * HD + ds * 32 + 16 * half;
        rB[et * 2 + ds] = ldpair(rr, rr + 8);
      }
    }
    v8f e[3];
#pragma unroll
    for (int et = 0; et < 3; ++et) {
      e[et] = vzero8();
      e[et] = wmma_bf16(aq[0], rB[et * 2 + 0], e[et]);
      e[et] = wmma_bf16(aq[1], rB[et * 2 + 1], e[et]);
    }

    // ---- skew gather + online softmax; stage P (bf16) in LDS ----
#pragma unroll
    for (int r = 0; r < 8; ++r) {
      const int i = r + 8 * half;
      const int jp0 = l15 - i + 15;                   // [0,30]
      const int src0 = (lane & 16) | (jp0 & 15);
      const float g00 = __shfl(e[0][r], src0, 32);
      const float g01 = __shfl(e[1][r], src0, 32);
      s0[r] += (jp0 < 16) ? g00 : g01;
      const int jp1 = l15 + 31 - i;                   // [16,46]
      const int src1 = (lane & 16) | (jp1 & 15);
      const float g11 = __shfl(e[1][r], src1, 32);
      const float g12 = __shfl(e[2][r], src1, 32);
      s1[r] += (jp1 < 32) ? g11 : g12;

      float mx = fmaxf(s0[r], s1[r]);
#pragma unroll
      for (int off = 8; off >= 1; off >>= 1) mx = fmaxf(mx, __shfl_xor(mx, off, 32));
      const float mnew = fmaxf(m[r], mx);
      const float alpha = __expf(m[r] - mnew);
      m[r] = mnew;
      const float p0 = __expf(s0[r] - mnew);
      const float p1 = __expf(s1[r] - mnew);
      float rs = p0 + p1;
#pragma unroll
      for (int off = 8; off >= 1; off >>= 1) rs += __shfl_xor(rs, off, 32);
      lsum[r] = lsum[r] * alpha + rs;
#pragma unroll
      for (int dt = 0; dt < 4; ++dt) o[dt][r] *= alpha;
      pb[i * 40 + l15]      = f2bf(p0);
      pb[i * 40 + 16 + l15] = f2bf(p1);
    }
    // ---- P as A-operand (16x32 bf16); O += P V (batch V tiles first) ----
    const u16* pr = pb + l15 * 40 + 16 * half;
    v16bf pA = ldpair(pr, pr + 8);
    v16bf vB[4];
#pragma unroll
    for (int dt = 0; dt < 4; ++dt) {
      const u16* vr = vbase + (size_t)(dt * 16 + l15) * Ls + k0 + 16 * half;
      vB[dt] = ldpair(vr, vr + 8);
    }
#pragma unroll
    for (int dt = 0; dt < 4; ++dt)
      o[dt] = wmma_bf16(pA, vB[dt], o[dt]);
  }
  // ---- normalize, store to [B*L, D] with col = h*64 + d (bf16) ----
#pragma unroll
  for (int r = 0; r < 8; ++r) {
    const float inv = 1.0f / lsum[r];
    const int row = b * Ls + l0 + r + 8 * half;
#pragma unroll
    for (int dt = 0; dt < 4; ++dt)
      attnb[(size_t)row * Dd + h * HD + dt * 16 + l15] = f2bf(o[dt][r] * inv);
  }
}

// ---------------- output projection: [4096,1024] x [1024,1024] + bias (f32 out) ----

__global__ __launch_bounds__(256) void gemm_proj_kernel(
    const u16* __restrict__ ab, const u16* __restrict__ wT,
    const float* __restrict__ bias, float* __restrict__ out) {
  alignas(16) __shared__ u16 bpanel[2][64 * 40];
  const int tid = threadIdx.x;
  const int wv = tid >> 5, lane = tid & 31, half = lane >> 4, l15 = lane & 15;
  const int m0 = (blockIdx.x / 16) * 128 + wv * 16;
  const int n0 = (blockIdx.x % 16) * 64;

  if (tid == 0)
    tdm_load_2d((unsigned)(__SIZE_TYPE__)&bpanel[0][0], wT + (size_t)n0 * Dd, 32, 64, Dd);

  v8f c[4];
#pragma unroll
  for (int i = 0; i < 4; ++i) c[i] = vzero8();

  const u16* arow = ab + (size_t)(m0 + l15) * Dd;
  const u16* ap0 = arow + 16 * half;
  v16bf a = ldpair(ap0, ap0 + 8);

  for (int k = 0, p = 0; k < Dd; k += 32, p ^= 1) {
    if (tid == 0) __builtin_amdgcn_s_wait_tensorcnt(0);
    __syncthreads();
    if (tid == 0 && k + 32 < Dd)
      tdm_load_2d((unsigned)(__SIZE_TYPE__)&bpanel[p ^ 1][0],
                  wT + (size_t)n0 * Dd + (k + 32), 32, 64, Dd);
    v16bf an = a;
    if (k + 32 < Dd) {
      const u16* ap = arow + (k + 32) + 16 * half;
      __builtin_prefetch(ap + 32, 0, 0);
      an = ldpair(ap, ap + 8);
    }
    const u16* bb = &bpanel[p][0];
    v16bf bt[4];
#pragma unroll
    for (int nt = 0; nt < 4; ++nt) {
      const u16* bp = bb + (nt * 16 + l15) * 40 + 16 * half;
      bt[nt] = ldpair(bp, bp + 8);
    }
#pragma unroll
    for (int nt = 0; nt < 4; ++nt)
      c[nt] = wmma_bf16(a, bt[nt], c[nt]);
    a = an;
  }
#pragma unroll
  for (int nt = 0; nt < 4; ++nt) {
#pragma unroll
    for (int r = 0; r < 8; ++r) {
      const int row = m0 + r + 8 * half;
      const int col = n0 + nt * 16 + l15;
      out[(size_t)row * Dd + col] = c[nt][r] + bias[col];
    }
  }
}

// ---------------- host-side orchestration ----------------

extern "C" void kernel_launch(void* const* d_in, const int* in_sizes, int n_in,
                              void* d_out, int out_size, void* d_ws, size_t ws_size,
                              hipStream_t stream) {
  (void)in_sizes; (void)n_in; (void)out_size; (void)ws_size;
  const float* x      = (const float*)d_in[0];  // [4,1024,1024]
  const float* W_qkv  = (const float*)d_in[1];  // [1024,3072]
  const float* W_proj = (const float*)d_in[2];  // [1024,1024]
  const float* b_proj = (const float*)d_in[3];  // [1024]
  const float* relpos = (const float*)d_in[4];  // [2047,64]
  float* out = (float*)d_out;

  char* ws = (char*)d_ws;
  u16* xb     = (u16*)(ws);                        // 8 MB   [4096,1024]
  u16* wqkvT  = (u16*)(ws +  8388608);             // 6 MB   [3072,1024]
  u16* wprojT = (u16*)(ws + 14680064);             // 2 MB   [1024,1024]
  u16* relb   = (u16*)(ws + 16777216);             // 256 KB [2048,64]
  u16* qb     = (u16*)(ws + 17039360);             // 8 MB   [B,H,L,HD]
  u16* kb     = (u16*)(ws + 25427968);             // 8 MB   [B,H,L,HD]
  u16* vtb    = (u16*)(ws + 33816576);             // 8 MB   [B,H,HD,L]
  u16* attnb  = (u16*)(ws + 42205184);             // 8 MB   [4096,1024]

  cvt_copy_kernel<<<(Bb * Ls * Dd) / 256, 256, 0, stream>>>(x, xb, Bb * Ls * Dd);
  cvt_transpose_kernel<<<(3 * Dd * Dd) / 256, 256, 0, stream>>>(W_qkv, wqkvT, Dd, 3 * Dd);
  cvt_transpose_kernel<<<(Dd * Dd) / 256, 256, 0, stream>>>(W_proj, wprojT, Dd, Dd);
  cvt_rel_kernel<<<(2048 * 64) / 256, 256, 0, stream>>>(relpos, relb);

  gemm_qkv_kernel<<<1536, 256, 0, stream>>>(xb, wqkvT, qb, kb, vtb);
  attn_kernel<<<1024, 128, 0, stream>>>(qb, kb, vtb, relb, attnb);
  gemm_proj_kernel<<<512, 256, 0, stream>>>(attnb, wprojT, b_proj, out);
}
